// SubbandGRUBlock_48962627174808
// MI455X (gfx1250) — matrix operand order repaired and checked by reading, not compiled, not run on hardware
//
#include <hip/hip_runtime.h>

typedef __attribute__((ext_vector_type(16))) _Float16 v16h;
typedef __attribute__((ext_vector_type(8)))  _Float16 v8h;
typedef __attribute__((ext_vector_type(8)))  float    v8f;
typedef __attribute__((ext_vector_type(4)))  unsigned int v4u;
typedef __attribute__((ext_vector_type(4)))  int          v4i;
typedef __attribute__((ext_vector_type(8)))  int          v8i_t;

#define T_STEPS 1000
#define BATCH   256
#define IN_DIM  128
#define HID     128
#define GATES   384      // 3*HID, gate order r,z,n
#define WPAD    136      // padded LDS row stride (f16): 272B rows, bank-conflict free
#define HGPAD   392      // padded LDS row stride (f32) for hg staging
#define XG_MT_PER_BLOCK 64
#define XG_BLOCKS ((T_STEPS * BATCH / 16) / XG_MT_PER_BLOCK)   // 250

#if __has_builtin(__builtin_amdgcn_tensor_load_to_lds)
#define HAVE_TDM 1
#endif

__device__ __forceinline__ float sigmoidf_fast(float x) {
    return 1.0f / (1.0f + __expf(-x));
}
__device__ __forceinline__ float tanhf_fast(float x) {
    return 1.0f - 2.0f / (__expf(2.0f * x) + 1.0f);   // saturates correctly
}
__device__ __forceinline__ v16h cat8(v8h lo, v8h hi) {
    return __builtin_shufflevector(lo, hi, 0,1,2,3,4,5,6,7,8,9,10,11,12,13,14,15);
}
__device__ __forceinline__ v8h cvt8(float4 a, float4 b) {
    v8h r;
    r[0]=(_Float16)a.x; r[1]=(_Float16)a.y; r[2]=(_Float16)a.z; r[3]=(_Float16)a.w;
    r[4]=(_Float16)b.x; r[5]=(_Float16)b.y; r[6]=(_Float16)b.z; r[7]=(_Float16)b.w;
    return r;
}
__device__ __forceinline__ unsigned int lds_off(const void* p) {
    // generic LDS pointer: addr[31:0] is the LDS byte offset (ISA aperture rules)
    return (unsigned int)(unsigned long long)p;
}

#ifdef HAVE_TDM
// TDM 2D tile load: dim0 elements (f16) per row, dim1 rows, tensor stride = dim0.
// Optional LDS padding (interval/amount codes per D# group1 spec).
__device__ __forceinline__ void tdm_load_2d(const void* gptr, unsigned int lds_byte,
                                            unsigned int dim0, unsigned int dim1,
                                            unsigned int pad_en,
                                            unsigned int pad_interval_code,
                                            unsigned int pad_amount_code)
{
    const unsigned long long ga = (unsigned long long)gptr;
    v4u g0;
    g0[0] = 1u;                                           // count=1, user descriptor
    g0[1] = lds_byte;                                     // lds_addr
    g0[2] = (unsigned int)ga;                             // global_addr[31:0]
    g0[3] = (unsigned int)((ga >> 32) & 0x01FFFFFFu)      // global_addr[56:32]
          | (2u << 30);                                   // type=2 ("image")
    v8i_t g1;
    g1[0] = (int)((1u << 16)                              // data_size=2 bytes
          | (pad_en << 20)
          | (pad_interval_code << 22)
          | (pad_amount_code << 25));
    g1[1] = (int)((dim0 & 0xFFFFu) << 16);                // tensor_dim0[15:0]
    g1[2] = (int)((dim0 >> 16) | ((dim1 & 0xFFFFu) << 16)); // tdim0[31:16] | tdim1[15:0]
    g1[3] = (int)((dim1 >> 16) | ((dim0 & 0xFFFFu) << 16)); // tdim1[31:16] | tile_dim0
    g1[4] = (int)(dim1 & 0xFFFFu);                        // tile_dim1 (tile_dim2=0)
    g1[5] = (int)dim0;                                    // tensor_dim0_stride[31:0]
    g1[6] = 0;
    g1[7] = 0;
    v4i z4 = {0, 0, 0, 0};
#if __clang_major__ >= 23
    v8i_t z8 = {0, 0, 0, 0, 0, 0, 0, 0};
    __builtin_amdgcn_tensor_load_to_lds(g0, g1, z4, z4, z8, 0);
#else
    __builtin_amdgcn_tensor_load_to_lds(g0, g1, z4, z4, 0);
#endif
}
#endif

// ---------------------------------------------------------------------------
// Kernel 0: convert both weight matrices f32 -> f16 into workspace.
// ---------------------------------------------------------------------------
__global__ void wcvt_kernel(const float* __restrict__ Wih, const float* __restrict__ Whh,
                            _Float16* __restrict__ Wih16, _Float16* __restrict__ Whh16) {
    int i = blockIdx.x * blockDim.x + threadIdx.x;
    if (i < GATES * IN_DIM) {
        Wih16[i] = (_Float16)Wih[i];
        Whh16[i] = (_Float16)Whh[i];
    }
}

// ---------------------------------------------------------------------------
// Kernel 1: xg = x @ W_ih^T + b_ih  (f16 out).  250 blocks x 64 M-tiles.
// Each wave owns one 16-row M-tile and sweeps all 24 gate tiles, so x is
// read exactly once. W_ih arrives in LDS via one TDM DMA with HW padding.
// ---------------------------------------------------------------------------
__global__ __launch_bounds__(256) void xgemm_kernel(
    const float* __restrict__ x, const _Float16* __restrict__ Wih16,
    const float* __restrict__ b_ih, _Float16* __restrict__ xg16)
{
    __shared__ __align__(16) _Float16 sW[GATES * WPAD];   // ~102 KB

    const int t = threadIdx.x;
#ifdef HAVE_TDM
    if (t < 32) {   // wave 0 issues the DMA; pad 4 dwords every 64 dwords -> stride 136
        tdm_load_2d(Wih16, lds_off(sW), IN_DIM, GATES, 1u, 5u, 3u);
        __builtin_amdgcn_s_wait_tensorcnt(0);
    }
#else
    for (int p = t; p < (GATES * IN_DIM) / 2; p += 256) {
        int row = p >> 6; int col = (p & 63) << 1;
        *(unsigned int*)&sW[row * WPAD + col] = ((const unsigned int*)Wih16)[p];
    }
#endif
    __syncthreads();

    const int lane   = t & 31;
    const int wv     = t >> 5;
    const int half   = (lane >> 4) & 1;
    const int khalfA = half ? 8 : 0;     // A layout: K split 8+8 per lane-half
    const int khalfB = half ? 8 : 0;     // B layout: K split 8+8 per lane-half (matches A)

    for (int it = 0; it < 8; ++it) {
        const int mt   = blockIdx.x * XG_MT_PER_BLOCK + it * 8 + wv;
        const int mrow = mt * 16 + (lane & 15);
        const float* xrow = x + (size_t)mrow * IN_DIM;

        v16h a[4];
        #pragma unroll
        for (int kb = 0; kb < 4; ++kb) {
            int k0 = kb * 32 + khalfA;
            float4 f0 = *(const float4*)(xrow + k0);
            float4 f1 = *(const float4*)(xrow + k0 + 4);
            float4 f2 = *(const float4*)(xrow + k0 + 16);
            float4 f3 = *(const float4*)(xrow + k0 + 20);
            a[kb] = cat8(cvt8(f0, f1), cvt8(f2, f3));
        }

        const int rbase = mt * 16 + (half ? 8 : 0);

        for (int grp = 0; grp < 8; ++grp) {   // 3 gate tiles in flight for ILP
            int nc[3];
            v8f acc[3];
            #pragma unroll
            for (int s = 0; s < 3; ++s) {
                nc[s] = (grp + s * 8) * 16 + (lane & 15);
                float bias = b_ih[nc[s]];
                #pragma unroll
                for (int i = 0; i < 8; ++i) acc[s][i] = bias;
            }
            #pragma unroll
            for (int kb = 0; kb < 4; ++kb) {
                int bk = kb * 32 + khalfB;
                #pragma unroll
                for (int s = 0; s < 3; ++s) {
                    v8h blo = *(const v8h*)&sW[nc[s] * WPAD + bk];
                    v8h bhi = *(const v8h*)&sW[nc[s] * WPAD + bk + 16];
                    acc[s] = __builtin_amdgcn_wmma_f32_16x16x32_f16(
                        false, a[kb], false, cat8(blo, bhi), (short)0, acc[s], false, false);
                }
            }
            #pragma unroll
            for (int s = 0; s < 3; ++s)
                #pragma unroll
                for (int r = 0; r < 8; ++r)
                    xg16[(size_t)(rbase + r) * GATES + nc[s]] = (_Float16)acc[s][r];
        }
    }
}

// ---------------------------------------------------------------------------
// Kernel 2: persistent GRU scan. 16 blocks, each owns a 16-row batch slice.
// W_hh f16 + h f16 + hg staging + double-buffered xg tile, all in LDS.
// TDM prefetches step t+1's xg slice while step t's WMMAs run.
// ---------------------------------------------------------------------------
__global__ __launch_bounds__(256) void gru_scan_kernel(
    const _Float16* __restrict__ xg16, const _Float16* __restrict__ Whh16,
    const float* __restrict__ b_hh, float* __restrict__ out)
{
    __shared__ __align__(16) _Float16 sW[GATES * WPAD];   // ~102 KB
    __shared__ __align__(16) _Float16 sH[16 * WPAD];      // h tile, f16
    __shared__ __align__(16) float    sHG[16 * HGPAD];    // hg staging, f32
#ifdef HAVE_TDM
    __shared__ __align__(16) _Float16 sXG[2][16 * GATES]; // xg double buffer, 2x12 KB
#endif

    const int t  = threadIdx.x;
    const int wv = t >> 5;
#ifdef HAVE_TDM
    const bool tdm_wave = (wv == 0);
    if (tdm_wave) {
        // in-order TDM: W first, then step-0 xg tile
        tdm_load_2d(Whh16, lds_off(sW), HID, GATES, 1u, 5u, 3u);
        tdm_load_2d(xg16 + (size_t)blockIdx.x * 16 * GATES,
                    lds_off(&sXG[0][0]), 16 * GATES, 1, 0u, 0u, 0u);
        __builtin_amdgcn_s_wait_tensorcnt(1);   // W complete; xg0 may still fly
    }
#else
    for (int p = t; p < (GATES * HID) / 2; p += 256) {
        int row = p >> 6; int col = (p & 63) << 1;
        *(unsigned int*)&sW[row * WPAD + col] = ((const unsigned int*)Whh16)[p];
    }
#endif
    for (int p = t; p < (16 * WPAD) / 2; p += 256)
        ((unsigned int*)sH)[p] = 0u;            // h0 = 0
    __syncthreads();

    const int lane   = t & 31;
    const int half   = (lane >> 4) & 1;
    const int khalfA = half ? 8 : 0;
    const int khalfB = half ? 8 : 0;
    const int arow   = lane & 15;
    const int crow   = half ? 8 : 0;

    int ncol[3]; float bias[3];
    #pragma unroll
    for (int s = 0; s < 3; ++s) {
        ncol[s] = (wv * 3 + s) * 16 + (lane & 15);
        bias[s] = b_hh[ncol[s]];
    }

    // gate-phase ownership: 8 consecutive h elements per thread
    const int b  = t >> 4;            // 0..15 local batch row
    const int jb = (t & 15) * 8;      // 0..120
    const int bg = blockIdx.x * 16 + b;
    float hreg[8];
    #pragma unroll
    for (int c = 0; c < 8; ++c) hreg[c] = 0.0f;

    const _Float16* xgp = xg16 + (size_t)bg * GATES;   // fallback path only
    float* outp = out + (size_t)bg * HID;

    for (int step = 0; step < T_STEPS; ++step) {
#ifdef HAVE_TDM
        if (tdm_wave && (step + 1 < T_STEPS)) {
            tdm_load_2d(xg16 + ((size_t)(step + 1) * BATCH + blockIdx.x * 16) * GATES,
                        lds_off(&sXG[(step + 1) & 1][0]), 16 * GATES, 1, 0u, 0u, 0u);
        }
#endif
        // ---- phase 1: hg = h @ W_hh^T + b_hh ----
        v16h amat[4];
        #pragma unroll
        for (int kb = 0; kb < 4; ++kb) {
            int k0 = kb * 32 + khalfA;
            v8h alo = *(const v8h*)&sH[arow * WPAD + k0];
            v8h ahi = *(const v8h*)&sH[arow * WPAD + k0 + 16];
            amat[kb] = cat8(alo, ahi);
        }
        #pragma unroll
        for (int s = 0; s < 3; ++s) {
            v8f acc;
            #pragma unroll
            for (int i = 0; i < 8; ++i) acc[i] = bias[s];
            #pragma unroll
            for (int kb = 0; kb < 4; ++kb) {
                int bk = kb * 32 + khalfB;
                v8h blo = *(const v8h*)&sW[ncol[s] * WPAD + bk];
                v8h bhi = *(const v8h*)&sW[ncol[s] * WPAD + bk + 16];
                acc = __builtin_amdgcn_wmma_f32_16x16x32_f16(
                    false, amat[kb], false, cat8(blo, bhi), (short)0, acc, false, false);
            }
            #pragma unroll
            for (int r = 0; r < 8; ++r)
                sHG[(crow + r) * HGPAD + ncol[s]] = acc[r];
        }
#ifdef HAVE_TDM
        if (tdm_wave) {   // make step's xg tile resident before the barrier
            if (step + 1 < T_STEPS) __builtin_amdgcn_s_wait_tensorcnt(1);
            else                    __builtin_amdgcn_s_wait_tensorcnt(0);
        }
#endif
        __syncthreads();

        // ---- phase 2: gates + h update ----
#ifdef HAVE_TDM
        const _Float16* xrow = &sXG[step & 1][b * GATES];
#else
        const _Float16* xrow = xgp;
        __builtin_prefetch(xrow + (size_t)BATCH * GATES, 0, 0);
        __builtin_prefetch(xrow + (size_t)BATCH * GATES + HID, 0, 0);
        __builtin_prefetch(xrow + (size_t)BATCH * GATES + 2 * HID, 0, 0);
#endif
        v8h xrv = *(const v8h*)(xrow + jb);
        v8h xzv = *(const v8h*)(xrow + HID + jb);
        v8h xnv = *(const v8h*)(xrow + 2 * HID + jb);

        const float* hg = &sHG[b * HGPAD];
        v8h hnew16;
        #pragma unroll
        for (int c = 0; c < 8; ++c) {
            float hr = hg[jb + c];
            float hz = hg[HID + jb + c];
            float hn = hg[2 * HID + jb + c];
            float r = sigmoidf_fast((float)xrv[c] + hr);
            float z = sigmoidf_fast((float)xzv[c] + hz);
            float n = tanhf_fast((float)xnv[c] + r * hn);
            float h = (1.0f - z) * n + z * hreg[c];
            hreg[c] = h;
            outp[jb + c] = h;
            hnew16[c] = (_Float16)h;
        }
        *(v8h*)&sH[b * WPAD + jb] = hnew16;

        xgp  += (size_t)BATCH * GATES;
        outp += (size_t)BATCH * HID;
        __syncthreads();
    }
}

// ---------------------------------------------------------------------------
// Kernel 3: LayerNorm over last dim (128), in place on d_out. Wave per row.
// ---------------------------------------------------------------------------
__global__ __launch_bounds__(256) void ln_kernel(float* __restrict__ out,
        const float* __restrict__ gamma, const float* __restrict__ beta)
{
    const int lane = threadIdx.x & 31;
    const int wv   = threadIdx.x >> 5;
    const size_t row = (size_t)blockIdx.x * 8 + wv;
    float* p = out + row * HID + lane * 4;
    float4 v = *(const float4*)p;
    float s  = v.x + v.y + v.z + v.w;
    float s2 = v.x*v.x + v.y*v.y + v.z*v.z + v.w*v.w;
    #pragma unroll
    for (int off = 16; off > 0; off >>= 1) {
        s  += __shfl_xor(s,  off, 32);
        s2 += __shfl_xor(s2, off, 32);
    }
    const float mean = s * (1.0f / 128.0f);
    const float var  = s2 * (1.0f / 128.0f) - mean * mean;
    const float rstd = rsqrtf(var + 1e-5f);
    const float4 g  = *(const float4*)(gamma + lane * 4);
    const float4 be = *(const float4*)(beta  + lane * 4);
    float4 y;
    y.x = (v.x - mean) * rstd * g.x + be.x;
    y.y = (v.y - mean) * rstd * g.y + be.y;
    y.z = (v.z - mean) * rstd * g.z + be.z;
    y.w = (v.w - mean) * rstd * g.w + be.w;
    *(float4*)p = y;
}

extern "C" void kernel_launch(void* const* d_in, const int* in_sizes, int n_in,
                              void* d_out, int out_size, void* d_ws, size_t ws_size,
                              hipStream_t stream) {
    const float* x     = (const float*)d_in[0];
    const float* W_ih  = (const float*)d_in[1];
    const float* W_hh  = (const float*)d_in[2];
    const float* b_ih  = (const float*)d_in[3];
    const float* b_hh  = (const float*)d_in[4];
    const float* gamma = (const float*)d_in[5];
    const float* beta  = (const float*)d_in[6];
    float* out = (float*)d_out;

    // workspace layout: xg f16 [T*B, 384] | W_ih f16 | W_hh f16  (~197 MB)
    _Float16* xg16  = (_Float16*)d_ws;
    const size_t xg_elems = (size_t)T_STEPS * BATCH * GATES;
    _Float16* Wih16 = xg16 + xg_elems;
    _Float16* Whh16 = Wih16 + (size_t)GATES * IN_DIM;

    wcvt_kernel<<<(GATES * IN_DIM + 255) / 256, 256, 0, stream>>>(W_ih, W_hh, Wih16, Whh16);
    xgemm_kernel<<<XG_BLOCKS, 256, 0, stream>>>(x, Wih16, b_ih, xg16);
    gru_scan_kernel<<<BATCH / 16, 256, 0, stream>>>(xg16, Whh16, b_hh, out);
    ln_kernel<<<(T_STEPS * BATCH) / 8, 256, 0, stream>>>(out, gamma, beta);
}
